// MultiHeadGraphAttention_7842610282646
// MI455X (gfx1250) — matrix operand order repaired and checked by reading, not compiled, not run on hardware
//
#include <hip/hip_runtime.h>

typedef __bf16 bf16;
typedef __attribute__((ext_vector_type(16))) __bf16 v16bf;
typedef __attribute__((ext_vector_type(8)))  float  v8f;

#define NN    4096
#define CD    256
#define HEADS 4
#define DH    64
#define MR    4096   // rows per head after flat view (== NN here)

__device__ __forceinline__ bf16 f2bf(float f) {
  unsigned u = __builtin_bit_cast(unsigned, f);
  unsigned r = u + 0x7FFFu + ((u >> 16) & 1u);
  unsigned short h = (unsigned short)(r >> 16);
  return __builtin_bit_cast(bf16, h);
}
__device__ __forceinline__ float bf2f(bf16 b) {
  unsigned short h = __builtin_bit_cast(unsigned short, b);
  unsigned u = (unsigned)h << 16;
  return __builtin_bit_cast(float, u);
}
__device__ __forceinline__ v8f wmma_bf16(v16bf a, v16bf b, v8f c) {
  return __builtin_amdgcn_wmma_f32_16x16x32_bf16(false, a, false, b, (short)0, c, false, false);
}
// A fragment (16x32 bf16): lane&15 = M row; K = (lane>=16?16:0) + e
__device__ __forceinline__ v16bf load_afrag(const bf16* __restrict__ row, int hi, int koff) {
  v16bf a; int base = koff + hi * 16;
#pragma unroll
  for (int e = 0; e < 8; ++e)  a[e] = row[base + e];
#pragma unroll
  for (int e = 8; e < 16; ++e) a[e] = row[base + e];
  return a;
}
// B fragment (32x16): lane&15 = N col; K = (lane>=16?16:0) + e  -> 16 contiguous when B rows are K-major
__device__ __forceinline__ v16bf load_bfrag(const bf16* __restrict__ rowk) {
  v16bf b;
#pragma unroll
  for (int e = 0; e < 16; ++e) b[e] = rowk[e];
  return b;
}
// CDNA5 async global->LDS copy, 16B per lane (ASYNCcnt tracked); low 32 bits of a
// generic pointer to __shared__ are the in-allocation LDS byte offset.
__device__ __forceinline__ void async_copy16(void* lds_dst, const void* gsrc) {
  unsigned loff = (unsigned)(size_t)lds_dst;
  asm volatile("global_load_async_to_lds_b128 %0, %1, off"
               :: "v"(loff), "v"(gsrc) : "memory");
}
__device__ __forceinline__ void wait_async0() {
  asm volatile("s_wait_asynccnt 0x0" ::: "memory");
}

// ---------------- utility kernels ----------------
__global__ void k_zero_f32(float* __restrict__ p, int n4) {
  int i = blockIdx.x * blockDim.x + threadIdx.x;
  if (i < n4) { float4 z; z.x = z.y = z.z = z.w = 0.f; ((float4*)p)[i] = z; }
}
__global__ void k_cvt_bf16(const float* __restrict__ in, bf16* __restrict__ out, int n) {
  int i = blockIdx.x * blockDim.x + threadIdx.x;
  if (i < n) out[i] = f2bf(in[i]);
}
// edge scatter into dense conv matrix (flat reinterpret of [N,N,2] as [2,N,N], 1x1 conv folded in)
__global__ void k_scatter(const int* __restrict__ ei, const float* __restrict__ sgn,
                          const float* __restrict__ conv_w, float* __restrict__ conv, int E) {
  int e = blockIdx.x * blockDim.x + threadIdx.x;
  if (e >= E) return;
  int s = ei[e], t = ei[E + e];
  float w0 = conv_w[0], w1 = conv_w[1];
  int thi = (t >= 2048) ? 1 : 0;
#pragma unroll
  for (int ch = 0; ch < 2; ++ch) {
    float val = sgn[2 * e + ch];
    int b = 2 * t + ch - thi * 4096;
    if (s < 2048) atomicAdd(conv + (size_t)(2 * s + thi) * NN + b, w0 * val);
    else          atomicAdd(conv + (size_t)(2 * (s - 2048) + thi) * NN + b, w1 * val);
  }
}
// per-64-chunk sum of squares: [N,4] layout == [H,4096]
__global__ void k_chunk_sq(const bf16* __restrict__ q, float* __restrict__ q2) {
  int i = blockIdx.x * blockDim.x + threadIdx.x;
  if (i >= NN * 4) return;
  const bf16* p = q + (size_t)i * 64;
  float s = 0.f;
#pragma unroll 8
  for (int j = 0; j < 64; ++j) { float f = bf2f(p[j]); s += f * f; }
  q2[i] = s;
}
// per-head transpose of V: vt[h][c][m] = Vh[m][c]  -> contiguous B-fragment rows for P@V
__global__ void k_transpose_v(const bf16* __restrict__ v, bf16* __restrict__ vt) {
  int i = blockIdx.x * blockDim.x + threadIdx.x;
  if (i >= NN * CD) return;
  int n = i >> 8, c = i & 255;
  int h = n >> 10;
  int m = ((n & 1023) << 2) + (c >> 6);
  int dc = c & 63;
  vt[(size_t)h * DH * MR + (size_t)dc * MR + m] = v[i];
}

// ---------------- WMMA GEMM: out[n,c] = sum_k A[n,k]*W[c,k], 64x64 tile / 128 thr ----------------
// Double-buffered: async copy of k-slice t+1 overlaps WMMA on slice t.
__global__ __launch_bounds__(128) void k_gemm64(const bf16* __restrict__ A, const bf16* __restrict__ W,
                                                float* __restrict__ outF, bf16* __restrict__ outB,
                                                int use_f32) {
  __shared__ alignas(16) bf16 As[2][64][32];
  __shared__ alignas(16) bf16 Bs[2][64][32];
  int r0 = blockIdx.x * 64, c0 = blockIdx.y * 64;
  int tid = threadIdx.x, lane = tid & 31, wid = tid >> 5;
  int m = lane & 15, hi = lane >> 4;
  v8f acc[4];
#pragma unroll
  for (int n = 0; n < 4; ++n)
#pragma unroll
    for (int v = 0; v < 8; ++v) acc[n][v] = 0.f;

  auto issue_k = [&](int k0, int b) {
    for (int u = tid; u < 256; u += 128) {
      int row = u >> 2, cc = u & 3;
      async_copy16((char*)&As[b][0][0] + (size_t)u * 16, A + (size_t)(r0 + row) * CD + k0 + cc * 8);
      async_copy16((char*)&Bs[b][0][0] + (size_t)u * 16, W + (size_t)(c0 + row) * CD + k0 + cc * 8);
    }
  };

  issue_k(0, 0);
  wait_async0();
  __syncthreads();
  int buf = 0;
  for (int k0 = 0; k0 < CD; k0 += 32) {
    if (k0 + 32 < CD) issue_k(k0 + 32, buf ^ 1);   // prefetch next slice (overlaps WMMA)
    v16bf a = load_afrag(&As[buf][wid * 16 + m][0], hi, 0);
#pragma unroll
    for (int n = 0; n < 4; ++n) {
      v16bf b = load_bfrag(&Bs[buf][16 * n + m][hi * 16]);
      acc[n] = wmma_bf16(a, b, acc[n]);
    }
    wait_async0();
    __syncthreads();
    buf ^= 1;
  }
#pragma unroll
  for (int n = 0; n < 4; ++n)
#pragma unroll
    for (int v = 0; v < 8; ++v) {
      int grow = r0 + wid * 16 + v + 8 * hi;
      int gcol = c0 + 16 * n + m;
      float val = acc[n][v];
      if (use_f32) outF[(size_t)grow * CD + gcol] = val;
      else         outB[(size_t)grow * CD + gcol] = f2bf(val);
    }
}

// ---------------- flash attention: 64-row block x 4 waves, online softmax ----------------
// Double-buffered K / V^T / conv tiles: async fill of column block j+1 overlaps block j compute.
__global__ __launch_bounds__(128) void k_flash(const bf16* __restrict__ qb, const bf16* __restrict__ kb,
                                               const bf16* __restrict__ vtb, const float* __restrict__ q2,
                                               const float* __restrict__ k2, const float* __restrict__ conv,
                                               const float* __restrict__ conv_b, const float* __restrict__ param,
                                               bf16* __restrict__ ao) {
  __shared__ alignas(16) bf16 Qs[64][64];
  __shared__ alignas(16) bf16 Ks[2][64][64];
  __shared__ alignas(16) bf16 Vts[2][64][64];  // transposed V tile: Vts[b][c][k] = V[j0+k][c]
  __shared__ alignas(16) bf16 Ps[4][16][64];
  __shared__ alignas(16) float Cs[2][64][64];  // conv tile
  __shared__ float k2s[2][64];

  int h = blockIdx.y, r0 = blockIdx.x * 64;
  int tid = threadIdx.x, lane = tid & 31, wid = tid >> 5;
  int m = lane & 15, hi = lane >> 4;

  const bf16* Qh  = qb  + (size_t)h * MR * DH;
  const bf16* Kh  = kb  + (size_t)h * MR * DH;
  const bf16* Vth = vtb + (size_t)h * DH * MR;

  float pcl = param[0]; pcl = (pcl < 1.f) ? 1.f : pcl;          // clamp_min(1.0)
  float addc = pcl * conv_b[0];
  const float inv_s = 1.f / 16.f;                                // 1/sqrt(C)

  auto issue_tiles = [&](int j0, int b) {
    const bf16* ksrc = Kh + (size_t)j0 * DH;
    for (int u = tid; u < 512; u += 128) {                       // K + V^T, async to LDS
      async_copy16((char*)&Ks[b][0][0] + (size_t)u * 16, ksrc + (size_t)u * 8);
      async_copy16((char*)&Vts[b][0][0] + (size_t)u * 16,
                   Vth + (size_t)(u >> 3) * MR + j0 + (size_t)(u & 7) * 8);
    }
    for (int u = tid; u < 1024; u += 128)                        // conv tile (f32), async
      async_copy16((char*)&Cs[b][0][0] + (size_t)u * 16,
                   conv + (size_t)(r0 + (u >> 4)) * NN + j0 + (size_t)(u & 15) * 4);
    if (tid < 64) k2s[b][tid] = k2[h * MR + j0 + tid];
  };

  for (int u = tid; u < 512; u += 128)                           // Q tile once, async
    async_copy16((char*)&Qs[0][0] + (size_t)u * 16, Qh + (size_t)r0 * DH + (size_t)u * 8);
  issue_tiles(0, 0);
  wait_async0();
  __syncthreads();
  v16bf qA[2];
  qA[0] = load_afrag(&Qs[wid * 16 + m][0], hi, 0);
  qA[1] = load_afrag(&Qs[wid * 16 + m][0], hi, 32);

  float q2r[8], rm[8], rl[8];
  v8f o[4];
#pragma unroll
  for (int v = 0; v < 8; ++v) {
    q2r[v] = q2[h * MR + r0 + wid * 16 + v + 8 * hi];
    rm[v] = -1e30f; rl[v] = 0.f;
#pragma unroll
    for (int n = 0; n < 4; ++n) o[n][v] = 0.f;
  }

  int buf = 0;
  for (int j0 = 0; j0 < MR; j0 += 64) {
    if (j0 + 64 < MR) issue_tiles(j0 + 64, buf ^ 1);   // prefetch next block (overlaps compute)

    float lg[4][8];
#pragma unroll
    for (int n = 0; n < 4; ++n) {
      v8f s;
#pragma unroll
      for (int v = 0; v < 8; ++v) s[v] = 0.f;
#pragma unroll
      for (int ks = 0; ks < 2; ++ks) {
        v16bf b = load_bfrag(&Ks[buf][16 * n + m][ks * 32 + hi * 16]);   // B = K^T
        s = wmma_bf16(qA[ks], b, s);
      }
#pragma unroll
      for (int v = 0; v < 8; ++v) {
        float cterm = Cs[buf][wid * 16 + v + 8 * hi][16 * n + m];
        lg[n][v] = (q2r[v] + k2s[buf][16 * n + m] - 2.f * s[v]) * inv_s + pcl * cterm + addc;
      }
    }
    // online softmax: rows live in 16-lane halves of the wave (C/D layout)
#pragma unroll
    for (int v = 0; v < 8; ++v) {
      float tmax = fmaxf(fmaxf(lg[0][v], lg[1][v]), fmaxf(lg[2][v], lg[3][v]));
      for (int off = 1; off < 16; off <<= 1) tmax = fmaxf(tmax, __shfl_xor(tmax, off, 32));
      float mnew = fmaxf(rm[v], tmax);
      float scale = __expf(rm[v] - mnew);
      float rs = 0.f;
#pragma unroll
      for (int n = 0; n < 4; ++n) { float pe = __expf(lg[n][v] - mnew); lg[n][v] = pe; rs += pe; }
      for (int off = 1; off < 16; off <<= 1) rs += __shfl_xor(rs, off, 32);
      rl[v] = rl[v] * scale + rs;
      rm[v] = mnew;
#pragma unroll
      for (int n = 0; n < 4; ++n) o[n][v] *= scale;
    }
    // relayout P (C/D layout -> A fragments) through LDS
#pragma unroll
    for (int n = 0; n < 4; ++n)
#pragma unroll
      for (int v = 0; v < 8; ++v)
        Ps[wid][v + 8 * hi][16 * n + m] = f2bf(lg[n][v]);
    __syncthreads();
    v16bf pA[2];
    pA[0] = load_afrag(&Ps[wid][m][0], hi, 0);
    pA[1] = load_afrag(&Ps[wid][m][0], hi, 32);
#pragma unroll
    for (int n = 0; n < 4; ++n)
#pragma unroll
      for (int ks = 0; ks < 2; ++ks) {
        v16bf b = load_bfrag(&Vts[buf][16 * n + m][ks * 32 + hi * 16]); // contiguous via V^T
        o[n] = wmma_bf16(pA[ks], b, o[n]);
      }

    wait_async0();          // next tiles landed (copy ran under the compute above)
    __syncthreads();
    buf ^= 1;
  }
#pragma unroll
  for (int v = 0; v < 8; ++v) {
    float inv = 1.f / rl[v];
    int row = r0 + wid * 16 + v + 8 * hi;
#pragma unroll
    for (int n = 0; n < 4; ++n)
      ao[(size_t)h * MR * DH + (size_t)row * DH + 16 * n + m] = f2bf(o[n][v] * inv);
  }
}

extern "C" void kernel_launch(void* const* d_in, const int* in_sizes, int n_in,
                              void* d_out, int out_size, void* d_ws, size_t ws_size,
                              hipStream_t stream) {
  (void)n_in; (void)out_size; (void)ws_size;
  const float* x      = (const float*)d_in[0];
  const int*   ei     = (const int*)d_in[1];
  const float* sgn    = (const float*)d_in[2];
  const float* wq     = (const float*)d_in[3];
  const float* wk     = (const float*)d_in[4];
  const float* wv     = (const float*)d_in[5];
  const float* wo     = (const float*)d_in[6];
  const float* conv_w = (const float*)d_in[7];
  const float* conv_b = (const float*)d_in[8];
  const float* param  = (const float*)d_in[9];
  int E = in_sizes[1] / 2;

  char* ws = (char*)d_ws;
  size_t off = 0;
  float* conv = (float*)(ws + off); off += (size_t)NN * NN * 4;     // 64 MB dense conv logits
  bf16* xb  = (bf16*)(ws + off); off += (size_t)NN * CD * 2;
  bf16* wqb = (bf16*)(ws + off); off += (size_t)CD * CD * 2;
  bf16* wkb = (bf16*)(ws + off); off += (size_t)CD * CD * 2;
  bf16* wvb = (bf16*)(ws + off); off += (size_t)CD * CD * 2;
  bf16* wob = (bf16*)(ws + off); off += (size_t)CD * CD * 2;
  bf16* qb  = (bf16*)(ws + off); off += (size_t)NN * CD * 2;
  bf16* kb  = (bf16*)(ws + off); off += (size_t)NN * CD * 2;
  bf16* vb  = (bf16*)(ws + off); off += (size_t)NN * CD * 2;
  bf16* vtb = (bf16*)(ws + off); off += (size_t)NN * CD * 2;
  float* q2 = (float*)(ws + off); off += (size_t)NN * 4 * 4;
  float* k2 = (float*)(ws + off); off += (size_t)NN * 4 * 4;
  bf16* aob = (bf16*)(ws + off); off += (size_t)NN * CD * 2;

  int nconv4 = NN * NN / 4;
  k_zero_f32<<<(nconv4 + 255) / 256, 256, 0, stream>>>(conv, nconv4);

  int nx = NN * CD, nw = CD * CD;
  k_cvt_bf16<<<(nx + 255) / 256, 256, 0, stream>>>(x, xb, nx);
  k_cvt_bf16<<<(nw + 255) / 256, 256, 0, stream>>>(wq, wqb, nw);
  k_cvt_bf16<<<(nw + 255) / 256, 256, 0, stream>>>(wk, wkb, nw);
  k_cvt_bf16<<<(nw + 255) / 256, 256, 0, stream>>>(wv, wvb, nw);
  k_cvt_bf16<<<(nw + 255) / 256, 256, 0, stream>>>(wo, wob, nw);

  k_scatter<<<(E + 255) / 256, 256, 0, stream>>>(ei, sgn, conv_w, conv, E);

  dim3 gg(NN / 64, CD / 64);
  k_gemm64<<<gg, 128, 0, stream>>>(xb, wqb, nullptr, qb, 0);
  k_gemm64<<<gg, 128, 0, stream>>>(xb, wkb, nullptr, kb, 0);
  k_gemm64<<<gg, 128, 0, stream>>>(xb, wvb, nullptr, vb, 0);

  k_chunk_sq<<<(NN * 4 + 255) / 256, 256, 0, stream>>>(qb, q2);
  k_chunk_sq<<<(NN * 4 + 255) / 256, 256, 0, stream>>>(kb, k2);
  k_transpose_v<<<(NN * CD + 255) / 256, 256, 0, stream>>>(vb, vtb);

  dim3 ga(MR / 64, HEADS);
  k_flash<<<ga, 128, 0, stream>>>(qb, kb, vtb, q2, k2, conv, conv_b, param, aob);

  k_gemm64<<<gg, 128, 0, stream>>>(aob, wob, (float*)d_out, nullptr, 1);
}
